// GNTO_QF_70196945485989
// MI455X (gfx1250) — compile-verified, matrix operand order checked by reading, not run on hardware
//
#include <hip/hip_runtime.h>
#include <stdint.h>

// ---------------------------------------------------------------------------
// GNN cost model (FeatureEmbed + GATConv + pool + head) for gfx1250.
// All large GEMMs run on v_wmma_f32_16x16x32_f16 (f16 A/B, f32 accum).
// Scatter/softmax parts use float atomics + encoded-uint atomicMax.
// ---------------------------------------------------------------------------

typedef _Float16 h8   __attribute__((ext_vector_type(8)));
typedef _Float16 v16h __attribute__((ext_vector_type(16)));
typedef float    v8f  __attribute__((ext_vector_type(8)));

#define AS_LD 40  // LDS row stride (f16 elems): 80B rows spread banks, keep 16B align

__device__ __forceinline__ float lrelu01(float x) { return x > 0.f ? x : 0.01f * x; }
__device__ __forceinline__ float lrelu02(float x) { return x > 0.f ? x : 0.2f * x; }

__device__ __forceinline__ unsigned fenc(float f) {
  unsigned u = __float_as_uint(f);
  return (u >> 31) ? ~u : (u | 0x80000000u);
}
__device__ __forceinline__ float fdec(unsigned u) {
  return __uint_as_float((u & 0x80000000u) ? (u & 0x7FFFFFFFu) : ~u);
}

union V16U { v16h v; h8 h[2]; };

// ---------------------------------------------------------------------------
// Generic WMMA GEMM: C[M x Nout] = A[M x K] * B[K x N], B pre-transposed+padded
// as Bt[Np x Kp] f16. Block tile 128(M) x 64(N), 8 waves (4x2), each wave does
// a 32x32 tile = 4 x v_wmma_f32_16x16x32_f16 per 32-wide k-step.
// EPI selects the fused epilogue. AF32: A is raw f32 with true K bound Ka.
// M must be a multiple of 128 (true for N=98304 and 3N=294912).
// ---------------------------------------------------------------------------
template <bool AF32, int EPI>
__global__ __launch_bounds__(256) void gemm_wmma(
    const void* __restrict__ Ap, int lda, int Ka,
    const _Float16* __restrict__ Bt, int Kp,
    void* __restrict__ Cp, int ldc,
    const float* __restrict__ p0, const float* __restrict__ p1,
    const int* __restrict__ p2) {
  __shared__ _Float16 As[128 * AS_LD];
  __shared__ _Float16 Bs[64 * AS_LD];
  const int tid = threadIdx.x;
  const int m0 = blockIdx.x * 128;
  const int n0 = blockIdx.y * 64;
  const int w = tid >> 5, lane = tid & 31;
  const int wm = w >> 1, wn = w & 1;
  const int lr = lane & 15, hi = lane >> 4;
  const int hk = hi * 8;

  const v8f vzero = {0.f, 0.f, 0.f, 0.f, 0.f, 0.f, 0.f, 0.f};
  v8f acc[2][2];
#pragma unroll
  for (int a = 0; a < 2; ++a)
#pragma unroll
    for (int b = 0; b < 2; ++b) acc[a][b] = vzero;

  const int ksteps = Kp >> 5;
  for (int ks = 0; ks < ksteps; ++ks) {
    const int k0 = ks << 5;
    if constexpr (AF32) {
      const float* A = (const float*)Ap;
#pragma unroll
      for (int i = 0; i < 16; ++i) {
        int idx = i * 256 + tid;            // 4096 elements of 128x32 tile
        int r = idx >> 5, kk = idx & 31;
        int gk = k0 + kk;
        float v = (gk < Ka) ? A[(size_t)(m0 + r) * (size_t)lda + gk] : 0.f;
        As[r * AS_LD + kk] = (_Float16)v;
      }
    } else {
      const _Float16* A = (const _Float16*)Ap;
#pragma unroll
      for (int i = 0; i < 4; ++i) {
        int c = i * 256 + tid;              // 1024 chunks of 8 f16 (16B)
        int r = c >> 2, kq = (c & 3) * 8;
        *(h8*)&As[r * AS_LD + kq] =
            *(const h8*)&A[(size_t)(m0 + r) * (size_t)lda + k0 + kq];
      }
    }
    {
      int c = tid;                          // 256 chunks: 64x32 B tile
      int r = c >> 2, kq = (c & 3) * 8;
      *(h8*)&Bs[r * AS_LD + kq] =
          *(const h8*)&Bt[(size_t)(n0 + r) * (size_t)Kp + k0 + kq];
    }
    __syncthreads();

    // A frag (16x32 f16): lane-half selects K 0..7/16..23 vs 8..15/24..31
    V16U af[2], bf[2];
#pragma unroll
    for (int mi = 0; mi < 2; ++mi) {
      int row = wm * 32 + mi * 16 + lr;
      af[mi].h[0] = *(const h8*)&As[row * AS_LD + hk];
      af[mi].h[1] = *(const h8*)&As[row * AS_LD + 16 + hk];
    }
#pragma unroll
    for (int ni = 0; ni < 2; ++ni) {
      int col = wn * 32 + ni * 16 + lr;
      bf[ni].h[0] = *(const h8*)&Bs[col * AS_LD + hk];
      bf[ni].h[1] = *(const h8*)&Bs[col * AS_LD + 16 + hk];
    }
#pragma unroll
    for (int mi = 0; mi < 2; ++mi)
#pragma unroll
      for (int ni = 0; ni < 2; ++ni)
        acc[mi][ni] = __builtin_amdgcn_wmma_f32_16x16x32_f16(
            false, af[mi].v, false, bf[ni].v, (short)0, acc[mi][ni], false, false);
    __syncthreads();
  }

  // ---- fused epilogues ----
#pragma unroll
  for (int mi = 0; mi < 2; ++mi) {
#pragma unroll
    for (int ni = 0; ni < 2; ++ni) {
#pragma unroll
      for (int v = 0; v < 8; ++v) {
        int grow = m0 + wm * 32 + mi * 16 + v + hi * 8;  // C: M = v + 8*(lane>>4)
        int gcol = n0 + wn * 32 + ni * 16 + lr;          // C: N = lane&15
        float a = acc[mi][ni][v];
        if constexpr (EPI == 1) {  // filter layer 1: lrelu(a+b), zero pad cols
          _Float16* C = (_Float16*)Cp;
          float val = 0.f;
          if (gcol < 73) val = lrelu01(a + p0[gcol]);
          C[(size_t)grow * ldc + gcol] = (_Float16)val;
        } else if constexpr (EPI == 2) {  // filter layer 2: lrelu * mask
          _Float16* C = (_Float16*)Cp;
          float val = 0.f;
          if (gcol < 73) val = lrelu01(a + p0[gcol]) * p1[grow];
          C[(size_t)grow * ldc + gcol] = (_Float16)val;
        } else if constexpr (EPI == 3) {  // hist: (a+b)*mask
          _Float16* C = (_Float16*)Cp;
          C[(size_t)grow * ldc + gcol] = (_Float16)((a + p0[gcol]) * p1[grow]);
        } else if constexpr (EPI == 4) {  // sample: a + b + table_embed[tid[row]]
          _Float16* C = (_Float16*)Cp;
          float val = a + p0[gcol] + p1[(size_t)p2[grow] * 64 + gcol];
          C[(size_t)grow * ldc + 201 + gcol] = (_Float16)val;
        } else if constexpr (EPI == 5) {  // proj: lrelu, zero-pad cols 329..351
          if (gcol < 352) {
            _Float16* C = (_Float16*)Cp;
            float val = 0.f;
            if (gcol < 329) val = lrelu01(a + p0[gcol]);
            C[(size_t)grow * ldc + gcol] = (_Float16)val;
          }
        } else if constexpr (EPI == 6) {  // GAT h: plain f32
          float* C = (float*)Cp;
          C[(size_t)grow * ldc + gcol] = a;
        }
      }
    }
  }
}

// --------------------------- prep / small kernels ---------------------------

// dst[Np x Kp] f16 = transpose+pad of src[K x Nsrc] f32 (x @ W layout).
__global__ void k_w2t(const float* __restrict__ src, _Float16* __restrict__ dst,
                      int K, int Nsrc, int Kp, int Np) {
  int t = blockIdx.x * 256 + threadIdx.x;
  if (t >= Np * Kp) return;
  int n = t / Kp, k = t % Kp;
  float v = (n < Nsrc && k < K) ? src[(size_t)k * Nsrc + n] : 0.f;
  dst[t] = (_Float16)v;
}

// fcin[3N x 128]: [col_emb(64) | op_emb(8) | val(1) | zeros]
__global__ void k_fcin(const int* __restrict__ col_id, const int* __restrict__ op_id,
                       const float* __restrict__ fval,
                       const float* __restrict__ col_emb,
                       const float* __restrict__ op_emb,
                       _Float16* __restrict__ fcin, int threeN) {
  int t = blockIdx.x * 256 + threadIdx.x;
  if (t >= threeN * 128) return;
  int row = t >> 7, c = t & 127;
  float v = 0.f;
  if (c < 64)        v = col_emb[(size_t)col_id[row] * 64 + c];
  else if (c < 72)   v = op_emb[(size_t)op_id[row] * 8 + (c - 64)];
  else if (c == 72)  v = fval[row];
  fcin[t] = (_Float16)v;
}

// h3[3N x 64]: h3[n*3+j][b] = hists[n][b*3+j] (b<50), zero pad
__global__ void k_h3(const float* __restrict__ hists, _Float16* __restrict__ h3,
                     int threeN) {
  int t = blockIdx.x * 256 + threadIdx.x;
  if (t >= threeN * 64) return;
  int row = t >> 6, b = t & 63;
  int n = row / 3, j = row - n * 3;
  float v = (b < 50) ? hists[(size_t)n * 150 + b * 3 + j] : 0.f;
  h3[t] = (_Float16)v;
}

// xcat base: type_emb (0:64), join_emb (137:201), zero pad (329:352)
__global__ void k_xbase(const int* __restrict__ type_id, const int* __restrict__ join_id,
                        const float* __restrict__ type_emb,
                        const float* __restrict__ join_emb,
                        _Float16* __restrict__ xcat, int N) {
  int t = blockIdx.x * 256 + threadIdx.x;
  if (t >= N * 352) return;
  int n = t / 352, c = t - n * 352;
  if (c < 64)                  xcat[t] = (_Float16)type_emb[(size_t)type_id[n] * 64 + c];
  else if (c >= 137 && c < 201) xcat[t] = (_Float16)join_emb[(size_t)join_id[n] * 64 + (c - 137)];
  else if (c >= 329)           xcat[t] = (_Float16)0.f;
}

// f_emb: sum over 3 filters / nfil -> xcat cols 64:137
__global__ void k_reduce_f(const _Float16* __restrict__ fc2, const float* __restrict__ mask,
                           _Float16* __restrict__ xcat, int N) {
  int t = blockIdx.x * 256 + threadIdx.x;
  if (t >= N * 73) return;
  int n = t / 73, c = t - n * 73;
  float nf = mask[n * 3] + mask[n * 3 + 1] + mask[n * 3 + 2];
  float s = (float)fc2[(size_t)(n * 3 + 0) * 128 + c] +
            (float)fc2[(size_t)(n * 3 + 1) * 128 + c] +
            (float)fc2[(size_t)(n * 3 + 2) * 128 + c];
  xcat[(size_t)n * 352 + 64 + c] = (_Float16)(s / nf);
}

// h_emb: sum over 3 / nfil -> xcat cols 265:329
__global__ void k_reduce_h(const _Float16* __restrict__ he, const float* __restrict__ mask,
                           _Float16* __restrict__ xcat, int N) {
  int t = blockIdx.x * 256 + threadIdx.x;
  if (t >= N * 64) return;
  int n = t >> 6, c = t & 63;
  float nf = mask[n * 3] + mask[n * 3 + 1] + mask[n * 3 + 2];
  float s = (float)he[(size_t)(n * 3 + 0) * 64 + c] +
            (float)he[(size_t)(n * 3 + 1) * 64 + c] +
            (float)he[(size_t)(n * 3 + 2) * 64 + c];
  xcat[(size_t)n * 352 + 265 + c] = (_Float16)(s / nf);
}

// attention logits per (node, head)
__global__ void k_att(const float* __restrict__ h, const float* __restrict__ att_src,
                      const float* __restrict__ att_dst, float* __restrict__ a_s,
                      float* __restrict__ a_d, int N) {
  int t = blockIdx.x * 256 + threadIdx.x;
  if (t >= N * 4) return;
  int n = t >> 2, hd = t & 3;
  const float* hp = h + (size_t)n * 256 + hd * 64;
  const float* ps = att_src + hd * 64;
  const float* pd = att_dst + hd * 64;
  float vs = 0.f, vd = 0.f;
#pragma unroll 8
  for (int c = 0; c < 64; ++c) {
    float hv = hp[c];
    vs += hv * ps[c];
    vd += hv * pd[c];
  }
  a_s[t] = vs;
  a_d[t] = vd;
}

__global__ void k_fill_u32(unsigned* __restrict__ p, unsigned v, int n) {
  int t = blockIdx.x * 256 + threadIdx.x;
  if (t < n) p[t] = v;
}

// segment max over dst via monotone-encoded uint atomicMax
__global__ void k_edge_max(const int* __restrict__ ei, const float* __restrict__ as_,
                           const float* __restrict__ ad_, unsigned* __restrict__ menc,
                           int N, int E) {
  int i = blockIdx.x * 256 + threadIdx.x;
  int T = E + N;
  if (i >= T) return;
  int s = (i < E) ? ei[i] : (i - E);
  int d = (i < E) ? ei[E + i] : (i - E);
#pragma unroll
  for (int hd = 0; hd < 4; ++hd) {
    float e = lrelu02(as_[s * 4 + hd] + ad_[d * 4 + hd]);
    atomicMax(&menc[d * 4 + hd], fenc(e));
  }
}

// ex = exp(e - m[dst]); den[dst] += ex
__global__ void k_edge_exp(const int* __restrict__ ei, const float* __restrict__ as_,
                           const float* __restrict__ ad_, const unsigned* __restrict__ menc,
                           float* __restrict__ den, float* __restrict__ ex, int N, int E) {
  int i = blockIdx.x * 256 + threadIdx.x;
  int T = E + N;
  if (i >= T) return;
  int s = (i < E) ? ei[i] : (i - E);
  int d = (i < E) ? ei[E + i] : (i - E);
#pragma unroll
  for (int hd = 0; hd < 4; ++hd) {
    float e = lrelu02(as_[s * 4 + hd] + ad_[d * 4 + hd]);
    float x = __expf(e - fdec(menc[d * 4 + hd]));
    ex[(size_t)i * 4 + hd] = x;
    atomicAdd(&den[d * 4 + hd], x);
  }
}

// out[dst] += alpha * h[src]; one thread per (edge, head*chan)
__global__ void k_agg(const int* __restrict__ ei, const float* __restrict__ ex,
                      const float* __restrict__ den, const float* __restrict__ h,
                      float* __restrict__ out, int N, int E) {
  int t = blockIdx.x * 256 + threadIdx.x;
  int T = (E + N) * 256;
  if (t >= T) return;
  int i = t >> 8, q = t & 255;
  int s = (i < E) ? ei[i] : (i - E);
  int d = (i < E) ? ei[E + i] : (i - E);
  int hd = q >> 6;
  float coef = ex[(size_t)i * 4 + hd] / den[d * 4 + hd];
  atomicAdd(&out[(size_t)d * 256 + q], coef * h[(size_t)s * 256 + q]);
}

// node out = relu(mean over heads + b); accumulate graph sums + counts
__global__ void k_node_out(const float* __restrict__ out, const float* __restrict__ b_gat,
                           const int* __restrict__ batch, float* __restrict__ gsum,
                           float* __restrict__ gcnt, int N) {
  int t = blockIdx.x * 256 + threadIdx.x;
  if (t >= N * 64) return;
  int n = t >> 6, c = t & 63;
  const float* o = out + (size_t)n * 256;
  float v = 0.25f * (o[c] + o[64 + c] + o[128 + c] + o[192 + c]) + b_gat[c];
  v = fmaxf(v, 0.f);
  int g = batch[n];
  atomicAdd(&gsum[(size_t)g * 64 + c], v);
  if (c == 0) atomicAdd(&gcnt[g], 1.f);
}

// head MLP + sigmoid: one 64-thread block per graph
__global__ void k_head(const float* __restrict__ gsum, const float* __restrict__ gcnt,
                       const float* __restrict__ W1, const float* __restrict__ b1,
                       const float* __restrict__ W2, const float* __restrict__ b2,
                       float* __restrict__ outp) {
  __shared__ float gm[64];
  __shared__ float red[64];
  int g = blockIdx.x, c = threadIdx.x;
  float cnt = fmaxf(gcnt[g], 1.f);
  gm[c] = gsum[(size_t)g * 64 + c] / cnt;
  __syncthreads();
  float acc = b1[c];
#pragma unroll 8
  for (int k = 0; k < 64; ++k) acc += gm[k] * W1[k * 64 + c];
  red[c] = fmaxf(acc, 0.f) * W2[c];
  __syncthreads();
  for (int s = 32; s > 0; s >>= 1) {
    if (c < s) red[c] += red[c + s];
    __syncthreads();
  }
  if (c == 0) outp[g] = 1.f / (1.f + __expf(-(red[0] + b2[0])));
}

// ---------------------------------------------------------------------------

static inline int cdiv(long long a, long long b) { return (int)((a + b - 1) / b); }

extern "C" void kernel_launch(void* const* d_in, const int* in_sizes, int n_in,
                              void* d_out, int out_size, void* d_ws, size_t ws_size,
                              hipStream_t stream) {
  (void)n_in; (void)ws_size;
  const int* type_id     = (const int*)d_in[0];
  const int* join_id     = (const int*)d_in[1];
  const int* filter_col  = (const int*)d_in[2];
  const int* filter_op   = (const int*)d_in[3];
  const float* filter_val  = (const float*)d_in[4];
  const float* filter_mask = (const float*)d_in[5];
  const float* hists       = (const float*)d_in[6];
  const int* table_id      = (const int*)d_in[7];
  const float* sample      = (const float*)d_in[8];
  const int* edge_index    = (const int*)d_in[9];
  const int* batch         = (const int*)d_in[10];
  const float* type_embed  = (const float*)d_in[11];
  const float* table_embed = (const float*)d_in[12];
  const float* column_embed= (const float*)d_in[13];
  const float* op_embed    = (const float*)d_in[14];
  const float* join_embed  = (const float*)d_in[15];
  const float* W_f1 = (const float*)d_in[16]; const float* b_f1 = (const float*)d_in[17];
  const float* W_f2 = (const float*)d_in[18]; const float* b_f2 = (const float*)d_in[19];
  const float* W_sample = (const float*)d_in[20]; const float* b_sample = (const float*)d_in[21];
  const float* W_hist = (const float*)d_in[22]; const float* b_hist = (const float*)d_in[23];
  const float* W_proj = (const float*)d_in[24]; const float* b_proj = (const float*)d_in[25];
  const float* W_gat = (const float*)d_in[26];
  const float* att_src = (const float*)d_in[27]; const float* att_dst = (const float*)d_in[28];
  const float* b_gat = (const float*)d_in[29];
  const float* W_h1 = (const float*)d_in[30]; const float* b_h1 = (const float*)d_in[31];
  const float* W_h2 = (const float*)d_in[32]; const float* b_h2 = (const float*)d_in[33];

  const int N = in_sizes[0];        // 98304 (multiple of 128)
  const int E = in_sizes[9] / 2;    // 196608
  const int G = out_size;           // 4096
  const int N3 = N * 3;             // 294912 (multiple of 128)

  // ---- workspace layout ----
  char* ws = (char*)d_ws;
  size_t off = 0;
  auto alloc = [&](size_t bytes) -> char* {
    char* p = ws + off;
    off += (bytes + 255) & ~(size_t)255;
    return p;
  };
  _Float16* wf1t   = (_Float16*)alloc((size_t)128 * 96 * 2);
  _Float16* wf2t   = (_Float16*)alloc((size_t)128 * 96 * 2);
  _Float16* whistt = (_Float16*)alloc((size_t)64 * 64 * 2);
  _Float16* wsamt  = (_Float16*)alloc((size_t)64 * 1024 * 2);
  _Float16* wprojt = (_Float16*)alloc((size_t)384 * 352 * 2);
  _Float16* wgatt  = (_Float16*)alloc((size_t)256 * 352 * 2);
  _Float16* xcat   = (_Float16*)alloc((size_t)N * 352 * 2);
  _Float16* xproj  = (_Float16*)alloc((size_t)N * 352 * 2);
  float*    gsum   = (float*)alloc((size_t)G * 64 * 4);
  float*    gcnt   = (float*)alloc((size_t)G * 4);
  char* tmp0 = ws + off;  // temp region: fcin/fc1/fc2, later reused for GAT bufs
  _Float16* fcin = (_Float16*)alloc((size_t)N3 * 128 * 2);
  _Float16* fc1  = (_Float16*)alloc((size_t)N3 * 128 * 2);
  _Float16* fc2  = (_Float16*)alloc((size_t)N3 * 128 * 2);
  _Float16* h3   = (_Float16*)alloc((size_t)N3 * 64 * 2);
  _Float16* he   = (_Float16*)alloc((size_t)N3 * 64 * 2);
  // Overlay (fc chain is dead once xcat is assembled): 212.3MB <= 226.5MB
  float*    hbuf  = (float*)tmp0;             // [N,256]
  float*    outbuf = hbuf + (size_t)N * 256;  // [N,256]
  float*    a_s   = outbuf + (size_t)N * 256; // [N,4]
  float*    a_d   = a_s + (size_t)N * 4;
  unsigned* menc  = (unsigned*)(a_d + (size_t)N * 4);
  float*    den   = (float*)(menc + (size_t)N * 4);
  float*    exb   = den + (size_t)N * 4;      // [(E+N),4]

  // ---- weight transpose/pad to f16 ----
  k_w2t<<<cdiv(128 * 96, 256), 256, 0, stream>>>(W_f1, wf1t, 73, 73, 96, 128);
  k_w2t<<<cdiv(128 * 96, 256), 256, 0, stream>>>(W_f2, wf2t, 73, 73, 96, 128);
  k_w2t<<<cdiv(64 * 64, 256), 256, 0, stream>>>(W_hist, whistt, 50, 64, 64, 64);
  k_w2t<<<cdiv(64 * 1024, 256), 256, 0, stream>>>(W_sample, wsamt, 1000, 64, 1024, 64);
  k_w2t<<<cdiv(384 * 352, 256), 256, 0, stream>>>(W_proj, wprojt, 329, 329, 352, 384);
  k_w2t<<<cdiv(256 * 352, 256), 256, 0, stream>>>(W_gat, wgatt, 329, 256, 352, 256);

  // ---- gather-type prep ----
  k_fcin<<<cdiv((long long)N3 * 128, 256), 256, 0, stream>>>(
      filter_col, filter_op, filter_val, column_embed, op_embed, fcin, N3);
  k_h3<<<cdiv((long long)N3 * 64, 256), 256, 0, stream>>>(hists, h3, N3);
  k_xbase<<<cdiv((long long)N * 352, 256), 256, 0, stream>>>(
      type_id, join_id, type_embed, join_embed, xcat, N);

  // ---- filter branch: two 73x73 layers as WMMA GEMMs over [3N,96] ----
  gemm_wmma<false, 1><<<dim3(N3 / 128, 2), 256, 0, stream>>>(
      fcin, 128, 0, wf1t, 96, fc1, 128, b_f1, nullptr, nullptr);
  gemm_wmma<false, 2><<<dim3(N3 / 128, 2), 256, 0, stream>>>(
      fc1, 128, 0, wf2t, 96, fc2, 128, b_f2, filter_mask, nullptr);
  k_reduce_f<<<cdiv((long long)N * 73, 256), 256, 0, stream>>>(fc2, filter_mask, xcat, N);

  // ---- hist branch ----
  gemm_wmma<false, 3><<<dim3(N3 / 128, 1), 256, 0, stream>>>(
      h3, 64, 0, whistt, 64, he, 64, b_hist, filter_mask, nullptr);
  k_reduce_h<<<cdiv((long long)N * 64, 256), 256, 0, stream>>>(he, filter_mask, xcat, N);

  // ---- sample GEMM (A f32, K=1000) fused with table_embed gather ----
  gemm_wmma<true, 4><<<dim3(N / 128, 1), 256, 0, stream>>>(
      sample, 1000, 1000, wsamt, 1024, xcat, 352, b_sample, table_embed, table_id);

  // ---- projection: [N,329]@[329,329] + lrelu ----
  gemm_wmma<false, 5><<<dim3(N / 128, 6), 256, 0, stream>>>(
      xcat, 352, 0, wprojt, 352, xproj, 352, b_proj, nullptr, nullptr);

  // ---- GAT linear: [N,329]@[329,256] -> h f32 ----
  gemm_wmma<false, 6><<<dim3(N / 128, 4), 256, 0, stream>>>(
      xproj, 352, 0, wgatt, 352, hbuf, 256, nullptr, nullptr, nullptr);

  // ---- attention / edge softmax / aggregation ----
  k_att<<<cdiv((long long)N * 4, 256), 256, 0, stream>>>(hbuf, att_src, att_dst, a_s, a_d, N);
  k_fill_u32<<<cdiv((long long)N * 4, 256), 256, 0, stream>>>(menc, 0x007FFFFFu, N * 4); // enc(-inf)
  hipMemsetAsync(den, 0, (size_t)N * 4 * 4, stream);
  hipMemsetAsync(outbuf, 0, (size_t)N * 256 * 4, stream);
  hipMemsetAsync(gsum, 0, (size_t)G * 64 * 4, stream);
  hipMemsetAsync(gcnt, 0, (size_t)G * 4, stream);
  k_edge_max<<<cdiv(E + N, 256), 256, 0, stream>>>(edge_index, a_s, a_d, menc, N, E);
  k_edge_exp<<<cdiv(E + N, 256), 256, 0, stream>>>(edge_index, a_s, a_d, menc, den, exb, N, E);
  k_agg<<<cdiv((long long)(E + N) * 256, 256), 256, 0, stream>>>(
      edge_index, exb, den, hbuf, outbuf, N, E);

  // ---- mean over heads + relu + graph pooling ----
  k_node_out<<<cdiv((long long)N * 64, 256), 256, 0, stream>>>(
      outbuf, b_gat, batch, gsum, gcnt, N);

  // ---- head MLP + sigmoid ----
  k_head<<<G, 64, 0, stream>>>(gsum, gcnt, W_h1, b_h1, W_h2, b_h2, (float*)d_out);
}
